// MLP_21500606284006
// MI455X (gfx1250) — matrix operand order repaired and checked by reading, not compiled, not run on hardware
//
#include <hip/hip_runtime.h>
#include <math.h>

#define N_TOK  1024
#define DMODEL 2048
#define FFN    5632
#define NEXP   8
#define RANK   16
#define LSCALE 2.0f   // alpha/rank = 32/16

typedef __attribute__((ext_vector_type(16))) __bf16 v16bf;
typedef __attribute__((ext_vector_type(8)))  __bf16 v8bf;
typedef __attribute__((ext_vector_type(4)))  __bf16 v4bf;
typedef __attribute__((ext_vector_type(8)))  float  v8f;

union FragBF { v16bf v; v8bf h[2]; };

__device__ __forceinline__ float dot16(const float* a, const float4* b4) {
  float s = 0.f;
#pragma unroll
  for (int i = 0; i < 4; ++i) {
    float4 b = b4[i];
    s += a[4*i+0]*b.x + a[4*i+1]*b.y + a[4*i+2]*b.z + a[4*i+3]*b.w;
  }
  return s;
}

// ---------------------------------------------------------------------------
// one-time fp32 -> bf16 conversion (grid-stride, float4 -> 4x bf16)
// ---------------------------------------------------------------------------
__global__ __launch_bounds__(256)
void cvt_f32_to_bf16(const float* __restrict__ in, __bf16* __restrict__ out, int n4) {
  int i = blockIdx.x * 256 + threadIdx.x;
  const int stride = gridDim.x * 256;
  for (; i < n4; i += stride) {
    const float4 v = ((const float4*)in)[i];
    v4bf o;
    o.x = (__bf16)v.x; o.y = (__bf16)v.y; o.z = (__bf16)v.z; o.w = (__bf16)v.w;
    ((v4bf*)out)[i] = o;
  }
}

// ---------------------------------------------------------------------------
// C[M,N] = A[M,K] @ B[N,K]^T    A,B bf16 / C fp32
// 256 threads = 8 waves; C-tile 128x128; K-tile 32 (one WMMA K)
// Double-buffered LDS filled with GLOBAL_LOAD_ASYNC_TO_LDS_B128 (ASYNCcnt),
// rows padded to 80B so 16-lane b128 fragment reads are bank-conflict-free.
// ---------------------------------------------------------------------------
#define LDS_ROW_B   80                       // 32 bf16 data + 16B pad
#define LDS_MAT_B   (128 * LDS_ROW_B)        // 10240 B per matrix tile
#define LDS_BUF_B   (2 * LDS_MAT_B)          // A + B per buffer

__global__ __launch_bounds__(256)
void gemm_nt_bf16_async(const __bf16* __restrict__ A, const __bf16* __restrict__ B,
                        float* __restrict__ C, int M, int N, int K) {
  __shared__ __bf16 smem[2][2][128][40];     // [buf][A/B][row][k] (40 = 80B row)

  const int tid  = threadIdx.x;
  const int lane = tid & 31;
  const int wave = tid >> 5;
  const int wM   = wave & 3;                 // 4 waves along M (32 rows each)
  const int wN   = wave >> 2;                // 2 waves along N (64 cols each)
  const int bM   = blockIdx.y * 128;
  const int bN   = blockIdx.x * 128;
  const int lo   = lane & 15;
  const int hi   = lane >> 4;

  // async-copy mapping: each thread moves 2x16B per matrix per K-step
  const int trow = tid >> 2;                 // 0..63 (and +64 for second issue)
  const int tcol = tid & 3;                  // 16B column within 64B row
  const uint32_t ldsBase = (uint32_t)(uintptr_t)&smem[0][0][0][0];
  const uint64_t aBase   = (uint64_t)(uintptr_t)A;
  const uint64_t bBase   = (uint64_t)(uintptr_t)B;
  const uint64_t rowAdv  = (uint64_t)K * 128;          // 64 rows * K * 2B

  auto issue = [&](int kk) {
    const uint32_t buf = (uint32_t)(kk & 1);
    const uint32_t lA  = ldsBase + buf * LDS_BUF_B + trow * LDS_ROW_B + tcol * 16;
    const uint32_t lB  = lA + LDS_MAT_B;
    const uint64_t gA  = aBase + ((uint64_t)(bM + trow) * K + (uint64_t)kk * 32) * 2 + tcol * 16;
    const uint64_t gB  = bBase + ((uint64_t)(bN + trow) * K + (uint64_t)kk * 32) * 2 + tcol * 16;
    asm volatile("global_load_async_to_lds_b128 %0, %1, off"
                 :: "v"(lA), "v"(gA) : "memory");
    asm volatile("global_load_async_to_lds_b128 %0, %1, off"
                 :: "v"((uint32_t)(lA + 64 * LDS_ROW_B)), "v"(gA + rowAdv) : "memory");
    asm volatile("global_load_async_to_lds_b128 %0, %1, off"
                 :: "v"(lB), "v"(gB) : "memory");
    asm volatile("global_load_async_to_lds_b128 %0, %1, off"
                 :: "v"((uint32_t)(lB + 64 * LDS_ROW_B)), "v"(gB + rowAdv) : "memory");
  };

  v8f acc[2][4] = {};
  const int steps = K >> 5;

  issue(0);
  for (int k = 0; k < steps; ++k) {
    asm volatile("s_wait_asynccnt 0x0" ::: "memory");  // my buf[k&1] fills done
    __syncthreads();                                   // all waves' fills done +
                                                       // all done reading buf[(k+1)&1]
    if (k + 1 < steps) issue(k + 1);                   // overlap next fill with WMMA

    const __bf16* As = &smem[k & 1][0][0][0];
    const __bf16* Bs = &smem[k & 1][1][0][0];

    // A frag (16x32): lanes 0-15 hold K0-7/K16-23, lanes 16-31 hold K8-15/K24-31
    FragBF a[2], b[4];
#pragma unroll
    for (int mi = 0; mi < 2; ++mi) {
      const __bf16* p = As + (wM * 32 + mi * 16 + lo) * 40 + hi * 8;
      a[mi].h[0] = *(const v8bf*)p;
      a[mi].h[1] = *(const v8bf*)(p + 16);
    }
    // B frag (32x16): lane = column, lanes 0-15 hold K0-7/K16-23, lanes 16-31 hold K8-15/K24-31
#pragma unroll
    for (int ni = 0; ni < 4; ++ni) {
      const __bf16* p = Bs + (wN * 64 + ni * 16 + lo) * 40 + hi * 8;
      b[ni].h[0] = *(const v8bf*)p;
      b[ni].h[1] = *(const v8bf*)(p + 16);
    }
#pragma unroll
    for (int mi = 0; mi < 2; ++mi)
#pragma unroll
      for (int ni = 0; ni < 4; ++ni)
        acc[mi][ni] = __builtin_amdgcn_wmma_f32_16x16x32_bf16(
            false, a[mi].v, false, b[ni].v, (short)0, acc[mi][ni], false, false);
  }

  // C layout: VGPR i, lane L -> row = i + (L>>4)*8, col = L&15
#pragma unroll
  for (int mi = 0; mi < 2; ++mi)
#pragma unroll
    for (int ni = 0; ni < 4; ++ni) {
      const int row0 = bM + wM * 32 + mi * 16 + hi * 8;
      const int col  = bN + wN * 64 + ni * 16 + lo;
#pragma unroll
      for (int i = 0; i < 8; ++i)
        C[(size_t)(row0 + i) * N + col] = acc[mi][ni][i];
    }
}

// ---------------------------------------------------------------------------
// Router: logits = x @ gate_w^T, softmax -> top-2 -> renormalized weights.
// p0/(p0+p1) = 1/(1+exp(l1-l0)) (softmax denominator cancels).
// ---------------------------------------------------------------------------
__global__ __launch_bounds__(256)
void router_kernel(const float* __restrict__ x, const float* __restrict__ gw,
                   float* __restrict__ logits, int* __restrict__ selE,
                   float* __restrict__ selW) {
  const int n = blockIdx.x, tid = threadIdx.x;
  __shared__ float red[NEXP];
  if (tid < NEXP) red[tid] = 0.f;
  __syncthreads();
  float acc[NEXP] = {};
  for (int d = tid; d < DMODEL; d += 256) {
    const float xv = x[(size_t)n * DMODEL + d];
#pragma unroll
    for (int e = 0; e < NEXP; ++e) acc[e] += xv * gw[(size_t)e * DMODEL + d];
  }
#pragma unroll
  for (int e = 0; e < NEXP; ++e) atomicAdd(&red[e], acc[e]);
  __syncthreads();
  if (tid == 0) {
    float l[NEXP];
#pragma unroll
    for (int e = 0; e < NEXP; ++e) { l[e] = red[e]; logits[n * NEXP + e] = l[e]; }
    int i0 = 0;
#pragma unroll
    for (int e = 1; e < NEXP; ++e) if (l[e] > l[i0]) i0 = e;
    int i1 = (i0 == 0) ? 1 : 0;
#pragma unroll
    for (int e = 0; e < NEXP; ++e) if (e != i0 && l[e] > l[i1]) i1 = e;
    const float w0 = 1.f / (1.f + __expf(l[i1] - l[i0]));
    selE[n * 2 + 0] = i0; selE[n * 2 + 1] = i1;
    selW[n * 2 + 0] = w0; selW[n * 2 + 1] = 1.f - w0;
  }
}

// ---------------------------------------------------------------------------
// l1[n,e,r] = x[n,:]·A1[e,r,:], l3 likewise. Block per token; x row in LDS;
// threads 0-127 -> l1, 128-255 -> l3 (wave-aligned split).
// ---------------------------------------------------------------------------
__global__ __launch_bounds__(256)
void lora_in_kernel(const float* __restrict__ x, const float* __restrict__ A1,
                    const float* __restrict__ A3, float* __restrict__ l1,
                    float* __restrict__ l3) {
  const int n = blockIdx.x, tid = threadIdx.x;
  __shared__ float xs[DMODEL];
  for (int d = tid; d < DMODEL; d += 256) xs[d] = x[(size_t)n * DMODEL + d];
  __syncthreads();
  const float* Am = (tid < 128) ? A1 : A3;
  float*       Lo = (tid < 128) ? l1 : l3;
  const int idx = tid & 127;                       // e*16 + r
  const float4* ap = (const float4*)(Am + (size_t)idx * DMODEL);
  const float4* xp = (const float4*)xs;
  float s = 0.f;
  for (int q = 0; q < DMODEL / 4; ++q) {
    float4 a = ap[q], b = xp[q];
    s += a.x * b.x + a.y * b.y + a.z * b.z + a.w * b.w;
  }
  Lo[(size_t)n * (NEXP * RANK) + idx] = s;
}

// ---------------------------------------------------------------------------
// Per token, for the 2 selected experts only:
//   h1 = base1 + 2*l1·B1[f],  h3 = base3 + 2*l3·B3[f],  s = silu(h1)*h3
//   s_mix[f] (bf16) = sum_k w_k s_k ; l2[k,r] = (sum_f s_k A2[e,r,f]) * w_k * 2
// ---------------------------------------------------------------------------
__global__ __launch_bounds__(256)
void fuse_swiglu_kernel(const float* __restrict__ base1, const float* __restrict__ base3,
                        const float* __restrict__ l1,   const float* __restrict__ l3,
                        const float* __restrict__ B1,   const float* __restrict__ B3,
                        const float* __restrict__ A2,
                        const int* __restrict__ selE,   const float* __restrict__ selW,
                        __bf16* __restrict__ smixbf, float* __restrict__ l2out) {
  const int n = blockIdx.x, tid = threadIdx.x;
  __shared__ float l1s[2][RANK], l3s[2][RANK], red[32];
  __shared__ int   eSh[2];
  __shared__ float wSh[2];
  if (tid < 2) { eSh[tid] = selE[n * 2 + tid]; wSh[tid] = selW[n * 2 + tid]; }
  __syncthreads();
  if (tid < 32) {
    const int k = tid >> 4, r = tid & 15;
    l1s[k][r] = l1[((size_t)n * NEXP + eSh[k]) * RANK + r];
    l3s[k][r] = l3[((size_t)n * NEXP + eSh[k]) * RANK + r];
    red[tid]  = 0.f;
  }
  __syncthreads();

  float acc2[2][RANK] = {};
  const int e0 = eSh[0], e1 = eSh[1];
  const float w0 = wSh[0], w1 = wSh[1];

  for (int f = tid; f < FFN; f += 256) {
    const float b1v = base1[(size_t)n * FFN + f];
    const float b3v = base3[(size_t)n * FFN + f];
    float sm = 0.f;
#pragma unroll
    for (int k = 0; k < 2; ++k) {
      const int   e = k ? e1 : e0;
      const float w = k ? w1 : w0;
      const float4* p1 = (const float4*)(B1 + ((size_t)e * FFN + f) * RANK);
      const float4* p3 = (const float4*)(B3 + ((size_t)e * FFN + f) * RANK);
      const float h1 = b1v + LSCALE * dot16(l1s[k], p1);
      const float h3 = b3v + LSCALE * dot16(l3s[k], p3);
      const float s  = (h1 / (1.f + __expf(-h1))) * h3;   // silu(h1)*h3
      sm += w * s;
      const float* a2p = A2 + (size_t)e * RANK * FFN + f; // A2[e,r,f], stride F
#pragma unroll
      for (int r = 0; r < RANK; ++r) acc2[k][r] += s * a2p[(size_t)r * FFN];
    }
    smixbf[(size_t)n * FFN + f] = (__bf16)sm;
  }
  __syncthreads();
#pragma unroll
  for (int k = 0; k < 2; ++k)
#pragma unroll
    for (int r = 0; r < RANK; ++r) atomicAdd(&red[k * RANK + r], acc2[k][r]);
  __syncthreads();
  if (tid < 32) {
    const int k = tid >> 4;
    l2out[(size_t)n * 32 + tid] = red[tid] * wSh[k] * LSCALE;
  }
}

// ---------------------------------------------------------------------------
// out[n,d] += sum_k l2[n,k,:]·B2[e_k,d,:]
// ---------------------------------------------------------------------------
__global__ __launch_bounds__(256)
void lora_out_kernel(const float* __restrict__ l2, const float* __restrict__ B2,
                     const int* __restrict__ selE, float* __restrict__ out) {
  const int n = blockIdx.x, tid = threadIdx.x;
  __shared__ float l2s[32];
  __shared__ int   eSh[2];
  if (tid < 32) l2s[tid] = l2[(size_t)n * 32 + tid];
  if (tid < 2)  eSh[tid] = selE[n * 2 + tid];
  __syncthreads();
  for (int d = tid; d < DMODEL; d += 256) {
    float acc = 0.f;
#pragma unroll
    for (int k = 0; k < 2; ++k) {
      const float4* p = (const float4*)(B2 + ((size_t)eSh[k] * DMODEL + d) * RANK);
      acc += dot16(&l2s[k * RANK], p);
    }
    out[(size_t)n * DMODEL + d] += acc;
  }
}

// ---------------------------------------------------------------------------
extern "C" void kernel_launch(void* const* d_in, const int* in_sizes, int n_in,
                              void* d_out, int out_size, void* d_ws, size_t ws_size,
                              hipStream_t stream) {
  (void)in_sizes; (void)n_in; (void)out_size; (void)ws_size;
  const float* x    = (const float*)d_in[0];
  const float* gate = (const float*)d_in[1];
  const float* W1   = (const float*)d_in[2];
  const float* W3   = (const float*)d_in[3];
  const float* W2   = (const float*)d_in[4];
  const float* A1   = (const float*)d_in[5];
  const float* B1   = (const float*)d_in[6];
  const float* A3   = (const float*)d_in[7];
  const float* B3   = (const float*)d_in[8];
  const float* A2   = (const float*)d_in[9];
  const float* B2   = (const float*)d_in[10];

  float* out    = (float*)d_out;
  float* logits = out + (size_t)N_TOK * DMODEL;

  // ---- workspace carve-out: fp32 region, then 16B-aligned bf16 region ----
  int*   selE  = (int*)d_ws;
  float* selW  = (float*)(selE + N_TOK * 2);
  float* l1    = selW + N_TOK * 2;
  float* l3    = l1 + (size_t)N_TOK * NEXP * RANK;
  float* l2    = l3 + (size_t)N_TOK * NEXP * RANK;
  float* base1 = l2 + (size_t)N_TOK * 2 * RANK;
  float* base3 = base1 + (size_t)N_TOK * FFN;
  __bf16* xbf    = (__bf16*)(base3 + (size_t)N_TOK * FFN);
  __bf16* W1bf   = xbf  + (size_t)N_TOK * DMODEL;
  __bf16* W3bf   = W1bf + (size_t)FFN * DMODEL;
  __bf16* W2bf   = W3bf + (size_t)FFN * DMODEL;
  __bf16* smixbf = W2bf + (size_t)DMODEL * FFN;

  // one-time precision conversion of GEMM operands (amortized across tile reuse)
  {
    const size_t nx = (size_t)N_TOK * DMODEL / 4, nw = (size_t)FFN * DMODEL / 4;
    int gx = (int)((nx + 255) / 256); if (gx > 8192) gx = 8192;
    int gw = (int)((nw + 255) / 256); if (gw > 8192) gw = 8192;
    cvt_f32_to_bf16<<<gx, 256, 0, stream>>>(x,  xbf,  (int)nx);
    cvt_f32_to_bf16<<<gw, 256, 0, stream>>>(W1, W1bf, (int)nw);
    cvt_f32_to_bf16<<<gw, 256, 0, stream>>>(W3, W3bf, (int)nw);
    cvt_f32_to_bf16<<<gw, 256, 0, stream>>>(W2, W2bf, (int)nw);
  }

  router_kernel<<<N_TOK, 256, 0, stream>>>(x, gate, logits, selE, selW);
  lora_in_kernel<<<N_TOK, 256, 0, stream>>>(x, A1, A3, l1, l3);
  gemm_nt_bf16_async<<<dim3(FFN / 128, N_TOK / 128), 256, 0, stream>>>(
      xbf, W1bf, base1, N_TOK, FFN, DMODEL);
  gemm_nt_bf16_async<<<dim3(FFN / 128, N_TOK / 128), 256, 0, stream>>>(
      xbf, W3bf, base3, N_TOK, FFN, DMODEL);
  fuse_swiglu_kernel<<<N_TOK, 256, 0, stream>>>(
      base1, base3, l1, l3, B1, B3, A2, selE, selW, smixbf, l2);
  gemm_nt_bf16_async<<<dim3(DMODEL / 128, N_TOK / 128), 256, 0, stream>>>(
      smixbf, W2bf, out, N_TOK, DMODEL, FFN);
  lora_out_kernel<<<N_TOK, 256, 0, stream>>>(l2, B2, selE, out);
}
